// implicit_mask_73778948211193
// MI455X (gfx1250) — matrix operand order repaired and checked by reading, not compiled, not run on hardware
//
#include <hip/hip_runtime.h>
#include <math.h>

// CDNA5 wave32 WMMA vector types
typedef __attribute__((ext_vector_type(16))) _Float16 v16h;
typedef __attribute__((ext_vector_type(8)))  float    v8f;

#define NLEV 8
#define TBL  65536   // hash table entries per level (F=2 floats each)

struct EncParams {
  float scale[NLEV];
  int   res[NLEV];
  int   hashed[NLEV];
};

__device__ __forceinline__ float sigmoidf_fast(float x) {
  return 1.0f / (1.0f + __expf(-x));
}

__global__ __launch_bounds__(256) void ngp_mask_fused_kernel(
    const float* __restrict__ uvi,     // [N,3]
    const float* __restrict__ table,   // [8,65536,2]
    const float* __restrict__ W1,      // [16,64] row-major
    const float* __restrict__ b1,      // [64]
    const float* __restrict__ W2,      // [64,1]
    const float* __restrict__ b2,      // [1]
    float* __restrict__ out,           // [N]
    EncParams ep)
{
  // 256 rows x 24 halfs (16 used + pad). Row stride 48B keeps 16B alignment
  // for the b128 A-tile reads while breaking the worst bank patterns.
  __shared__ _Float16 encLds[256][24];

  const int tid = threadIdx.x;
  const int gpt = blockIdx.x * 256 + tid;   // grid is exact: N % 256 == 0

  // ---------------- Phase 1: hash-grid encode, one point per thread ----------
  const float ux = uvi[3 * gpt + 0];
  const float uy = uvi[3 * gpt + 1];
  const float uz = uvi[3 * gpt + 2];

  unsigned int* encRow = (unsigned int*)(&encLds[tid][0]);

#pragma unroll
  for (int l = 0; l < NLEV; ++l) {
    const float        s   = ep.scale[l];
    const unsigned int r   = (unsigned int)ep.res[l];
    const unsigned int rr  = r * r;
    // Branchless select mask: all-ones if this level uses the spatial hash.
    const unsigned int hm  = ep.hashed[l] ? 0xFFFFFFFFu : 0u;

    const float px = ux * s + 0.5f;
    const float py = uy * s + 0.5f;
    const float pz = uz * s + 0.5f;
    const float fxp = floorf(px), fyp = floorf(py), fzp = floorf(pz);
    const float wx = px - fxp, wy = py - fyp, wz = pz - fzp;
    const unsigned int x0 = (unsigned int)fxp;
    const unsigned int y0 = (unsigned int)fyp;
    const unsigned int z0 = (unsigned int)fzp;

    const float* tl = table + ((size_t)l << 17);   // l * T * F

    float a0 = 0.0f, a1 = 0.0f;
#pragma unroll
    for (int c = 0; c < 8; ++c) {
      const unsigned int cx = x0 + (c & 1);
      const unsigned int cy = y0 + ((c >> 1) & 1);
      const unsigned int cz = z0 + ((c >> 2) & 1);
      // Compute both index flavors; select without control flow. The extra
      // VALU is free under gather latency and removes ~16 uniform branches.
      const unsigned int idxh =
          (cx ^ (cy * 2654435761u) ^ (cz * 805459861u)) & (TBL - 1);
      const unsigned int idxd = cx + cy * r + cz * rr;
      const unsigned int idx  = (idxh & hm) | (idxd & ~hm);

      const float w = ((c & 1)        ? wx : 1.0f - wx)
                    * (((c >> 1) & 1) ? wy : 1.0f - wy)
                    * (((c >> 2) & 1) ? wz : 1.0f - wz);
      const float2 t2 = *(const float2*)(tl + 2u * idx);  // global_load_b64 gather
      a0 = fmaf(t2.x, w, a0);
      a1 = fmaf(t2.y, w, a1);
    }
    union { _Float16 h[2]; unsigned int u; } pk;
    pk.h[0] = (_Float16)a0;
    pk.h[1] = (_Float16)a1;
    encRow[l] = pk.u;   // halfs 2l, 2l+1 of this point's feature row
  }

  __syncthreads();

  // ---------------- Phase 2: MLP via WMMA, 32 points per wave -----------------
  const int lane = tid & 31;
  const int wv   = tid >> 5;
  const int col  = lane & 15;          // N-column this lane owns in B/D tiles

  // Four 16(K)x16(N) B tiles of W1 in f16. Per documented 16-bit B layout for
  // 32x16: lanes 0-15 hold K=0..7 (halves 0..7), lanes 16-31 hold K=8..15;
  // halves 8..15 (K=16..31) are our zero padding.
  v16h  Bt[4];
  float b1n[4], w2n[4];
#pragma unroll
  for (int blk = 0; blk < 4; ++blk) {
    const int n = blk * 16 + col;
    v16h b = {};
    {
#pragma unroll
      for (int k = 0; k < 8; ++k) b[k] = (_Float16)W1[(8 * (lane >> 4) + k) * 64 + n];
    }
    Bt[blk]  = b;
    b1n[blk] = b1[n];
    w2n[blk] = W2[n];
  }
  const float bias2 = b2[0];

  const int wbase  = wv * 32;                  // first point of this wave (in block)
  const int k0byte = (lane < 16) ? 0 : 16;     // A layout: K0..7 vs K8..15 halves

#pragma unroll
  for (int p = 0; p < 2; ++p) {
    // A tile: M = 16 points, K = 16 features zero-padded to 32.
    const int mrow = wbase + p * 16 + col;     // LDS row for this lane's M row
    union { v16h h; unsigned int u[8]; } a;
#pragma unroll
    for (int j = 0; j < 8; ++j) a.u[j] = 0u;   // halves 8..15 => K=16..31 pad
    const unsigned int* src =
        (const unsigned int*)((const char*)(&encLds[mrow][0]) + k0byte);
    a.u[0] = src[0]; a.u[1] = src[1]; a.u[2] = src[2]; a.u[3] = src[3];

    float part[8];
#pragma unroll
    for (int v = 0; v < 8; ++v) part[v] = 0.0f;

#pragma unroll
    for (int blk = 0; blk < 4; ++blk) {
      v8f c = {};
      v8f d = __builtin_amdgcn_wmma_f32_16x16x32_f16(
          /*neg_a=*/false, a.h, /*neg_b=*/false, Bt[blk],
          /*c_mod=*/(short)0, c, /*reuse_a=*/false, /*reuse_b=*/false);
      // Layer 2 folded directly on the D layout: this lane's column is hidden
      // unit n = blk*16+col, rows M = v + 8*(lane>=16).
#pragma unroll
      for (int v = 0; v < 8; ++v) {
        float h = d[v] + b1n[blk];
        h = fmaxf(h, 0.0f);
        part[v] = fmaf(h, w2n[blk], part[v]);
      }
    }

    // Reduce over the 16 lanes of each half-wave (masks < 16 never cross the
    // M-split between lanes 0-15 and 16-31).
#pragma unroll
    for (int m = 1; m < 16; m <<= 1) {
#pragma unroll
      for (int v = 0; v < 8; ++v) part[v] += __shfl_xor(part[v], m, 32);
    }

    if ((lane & 15) == 0) {
      const int mb = blockIdx.x * 256 + wbase + p * 16 + ((lane >> 4) << 3);
      float4 o0 = make_float4(sigmoidf_fast(part[0] + bias2),
                              sigmoidf_fast(part[1] + bias2),
                              sigmoidf_fast(part[2] + bias2),
                              sigmoidf_fast(part[3] + bias2));
      float4 o1 = make_float4(sigmoidf_fast(part[4] + bias2),
                              sigmoidf_fast(part[5] + bias2),
                              sigmoidf_fast(part[6] + bias2),
                              sigmoidf_fast(part[7] + bias2));
      *(float4*)(out + mb)     = o0;
      *(float4*)(out + mb + 4) = o1;
    }
  }
}

extern "C" void kernel_launch(void* const* d_in, const int* in_sizes, int n_in,
                              void* d_out, int out_size, void* d_ws, size_t ws_size,
                              hipStream_t stream) {
  const float* uvi   = (const float*)d_in[0];
  const float* table = (const float*)d_in[1];
  const float* W1    = (const float*)d_in[2];
  const float* b1    = (const float*)d_in[3];
  const float* W2    = (const float*)d_in[4];
  const float* b2    = (const float*)d_in[5];
  float*       out   = (float*)d_out;

  const int npts = in_sizes[0] / 3;   // 2,097,152

  // Replicate the reference's double-precision level-scale derivation exactly
  // (level 1/7 resolutions are sensitive to the last ulp of B_GROWTH).
  EncParams ep;
  const double Bg = exp(log(2048.0 / 16.0) / 7.0);
  for (int l = 0; l < NLEV; ++l) {
    const double s = 16.0 * pow(Bg, (double)l) - 1.0;
    ep.scale[l] = (float)s;
    const long long r = (long long)ceil(s) + 1;
    ep.res[l]    = (int)r;
    ep.hashed[l] = (r * r * r > (long long)TBL) ? 1 : 0;
  }

  const int blocks = npts / 256;      // N is an exact multiple of 256
  ngp_mask_fused_kernel<<<blocks, 256, 0, stream>>>(
      uvi, table, W1, b1, W2, b2, out, ep);
}
